// LetterDetectionRCNN_58377195487355
// MI455X (gfx1250) — matrix operand order repaired and checked by reading, not compiled, not run on hardware
//
#include <hip/hip_runtime.h>
#include <hip/hip_bf16.h>
#include <stdint.h>

typedef __attribute__((ext_vector_type(16))) __bf16 v16bf;
typedef __attribute__((ext_vector_type(8)))  float  v8f;
typedef unsigned short u16;
typedef unsigned int   u32;

// ---------- bf16 helpers (round-to-nearest-even) ----------
__device__ __forceinline__ u16 f2bf(float f) {
  u32 u = __float_as_uint(f);
  u32 r = u + 0x7FFFu + ((u >> 16) & 1u);
  return (u16)(r >> 16);
}
__device__ __forceinline__ float bf2f(u16 h) { return __uint_as_float(((u32)h) << 16); }
__device__ __forceinline__ __bf16 usbf(u16 h) { union { u16 s; __bf16 b; } x; x.s = h; return x.b; }

// ---------- weight repack: f32 OIHW -> bf16 [Opad][kk][Cinpad], zero padded ----------
__global__ void pack_weight_bf16(const float* __restrict__ w, u16* __restrict__ out,
                                 int Cout, int Cin, int kk, int Cinpad, int total) {
  int i = blockIdx.x * blockDim.x + threadIdx.x;
  if (i >= total) return;
  int Kpad = kk * Cinpad;
  int o = i / Kpad; int r = i - o * Kpad;
  int kp = r / Cinpad; int ci = r - kp * Cinpad;
  u16 v = 0;
  if (o < Cout && ci < Cin) v = f2bf(w[((size_t)o * Cin + ci) * kk + kp]);
  out[i] = v;
}

// ---------- input image: NCHW f32 -> NHWC bf16 ----------
__global__ void cvt_nchw_to_nhwc_bf16(const float* __restrict__ in, u16* __restrict__ out,
                                      int C, int HW, int total) {
  int i = blockIdx.x * blockDim.x + threadIdx.x;
  if (i >= total) return;
  int c = i % C; int t = i / C; int p = t % HW; int b = t / HW;
  out[i] = f2bf(in[((size_t)b * C + c) * HW + p]);
}

// ---------- generic implicit-GEMM conv via V_WMMA_F32_16X16X32_BF16 ----------
// NHWC activations. Wave tile: 64 Cout x 32 pixels = 4 M-subtiles x 2 N-groups;
// each A fragment feeds 2 WMMAs, each B fragment feeds 4 -> 1.5 wide loads/WMMA.
// Block = 4 waves = 128 pixels, same Cout range (A reuse via L2).
__global__ void __launch_bounds__(128)
conv_wmma_bf16(const u16* __restrict__ in, const u16* __restrict__ wgt,
               const float* __restrict__ scale, const float* __restrict__ shift,
               const u16* __restrict__ resid, int relu,
               u16* __restrict__ out_bf, float* __restrict__ out_f32,
               long long fbase, long long fbstride,
               int Cin, int H, int W, int Cout, int k, int stride, int pad,
               int OH, int OW, int Cinpad) {
  const int lane = threadIdx.x & 31;
  const int wv   = threadIdx.x >> 5;
  const int hh   = lane >> 4;    // half-wave select
  const int ln   = lane & 15;
  const int OHW  = OH * OW;
  const int pBase = blockIdx.x * 128 + wv * 32;
  if (pBase >= OHW) return;      // wave-uniform
  const int cBase = blockIdx.y * 64;
  const int b     = blockIdx.z;

  int pixg[2], iy0g[2], ix0g[2];
#pragma unroll
  for (int g = 0; g < 2; g++) {
    int p = pBase + 16 * g + ln;
    pixg[g] = p;
    int ps = (p < OHW) ? p : (OHW - 1);
    int oyv = ps / OW, oxv = ps - oyv * OW;
    iy0g[g] = oyv * stride - pad;
    ix0g[g] = oxv * stride - pad;
  }
  const int kk  = k * k;
  const long long HWs = (long long)H * W;
  const long long inB = (long long)b * HWs;
  const size_t Kpad = (size_t)kk * Cinpad;
  const u16* wrow0 = wgt + (size_t)(cBase + ln) * Kpad;   // A row for subtile 0

  v8f acc[2][4] = {{{}, {}, {}, {}}, {{}, {}, {}, {}}};

  auto mma8 = [&](const u16* arow, int c0, v16bf B0, v16bf B1) {
#pragma unroll
    for (int mt = 0; mt < 4; mt++) {
      const u32* wm = (const u32*)(arow + (size_t)mt * 16 * Kpad + c0);
      v16bf A;
#pragma unroll
      for (int j = 0; j < 8; j++) {
        int k0 = 2 * j + 16 * hh;
        u32 d = wm[k0 >> 1];
        A[2 * j]     = usbf((u16)(d & 0xFFFFu));
        A[2 * j + 1] = usbf((u16)(d >> 16));
      }
      acc[0][mt] = __builtin_amdgcn_wmma_f32_16x16x32_bf16(false, A, false, B0, (short)0,
                                                           acc[0][mt], false, false);
      acc[1][mt] = __builtin_amdgcn_wmma_f32_16x16x32_bf16(false, A, false, B1, (short)0,
                                                           acc[1][mt], false, false);
    }
  };

  if ((Cin & 31) == 0) {
    // ---- fast path: contiguous, 32B-aligned NHWC channel runs ----
    int ky = 0, kx = 0;
    for (int kp = 0; kp < kk; kp++) {
      bool ib[2]; long long pb[2];
#pragma unroll
      for (int g = 0; g < 2; g++) {
        int iy = iy0g[g] + ky, ix = ix0g[g] + kx;
        ib[g] = ((unsigned)iy < (unsigned)H) & ((unsigned)ix < (unsigned)W);
        pb[g] = ib[g] ? ((inB + (long long)iy * W + ix) * Cin) : 0;
      }
      const u16* arow = wrow0 + (size_t)kp * Cinpad;
      for (int c0 = 0; c0 < Cinpad; c0 += 32) {
        v16bf B0 = {}, B1 = {};
        if (ib[0]) B0 = *(const v16bf*)(in + pb[0] + 16 * hh + c0);   // 2x b128
        if (ib[1]) B1 = *(const v16bf*)(in + pb[1] + 16 * hh + c0);   // 2x b128
        mma8(arow, c0, B0, B1);
      }
      if (++kx == k) { kx = 0; ++ky; }
    }
  } else {
    // ---- masked path (stem, Cin=3): scalar gather with channel-tail mask ----
    int ky = 0, kx = 0;
    for (int kp = 0; kp < kk; kp++) {
      bool ib[2]; long long pb[2];
#pragma unroll
      for (int g = 0; g < 2; g++) {
        int iy = iy0g[g] + ky, ix = ix0g[g] + kx;
        ib[g] = ((unsigned)iy < (unsigned)H) & ((unsigned)ix < (unsigned)W);
        pb[g] = ib[g] ? ((inB + (long long)iy * W + ix) * Cin) : 0;
      }
      const u16* arow = wrow0 + (size_t)kp * Cinpad;
      for (int c0 = 0; c0 < Cinpad; c0 += 32) {
        const int climit = Cin - c0;
        v16bf Bg[2];
#pragma unroll
        for (int g = 0; g < 2; g++)
#pragma unroll
          for (int j = 0; j < 8; j++)
#pragma unroll
            for (int t = 0; t < 2; t++) {
              int kidx = 16 * hh + 2 * j + t;
              u16 v = 0;
              if (ib[g] && kidx < climit) v = in[pb[g] + c0 + kidx];
              Bg[g][2 * j + t] = usbf(v);
            }
        mma8(arow, c0, Bg[0], Bg[1]);
      }
      if (++kx == k) { kx = 0; ++ky; }
    }
  }

  const long long oB = (long long)b * OHW;
#pragma unroll
  for (int g = 0; g < 2; g++) {
    if (pixg[g] >= OHW) continue;        // post-WMMA, per-lane ok
    const long long obase = (oB + pixg[g]) * Cout;   // NHWC out pixel base
#pragma unroll
    for (int mt = 0; mt < 4; mt++) {
#pragma unroll
      for (int j = 0; j < 8; j++) {
        int co = cBase + mt * 16 + j + 8 * hh;   // D: VGPR j -> M = j + 8*half
        if (co >= Cout) continue;
        float v = acc[g][mt][j];
        if (scale) v *= scale[co];
        if (shift) v += shift[co];
        if (resid) v += bf2f(resid[obase + co]);
        if (relu)  v = fmaxf(v, 0.f);
        if (out_bf)  out_bf[obase + co] = f2bf(v);
        if (out_f32)
          out_f32[fbase + (long long)b * fbstride + (long long)co * OHW + pixg[g]] = v;
      }
    }
  }
}

// ---------- 3x3 stride-2 pad-1 maxpool (bf16, NHWC) ----------
__global__ void maxpool3s2(const u16* __restrict__ in, u16* __restrict__ out,
                           int C, int H, int W, int OH, int OW, int N) {
  int i = blockIdx.x * blockDim.x + threadIdx.x;
  int total = N * C * OH * OW;
  if (i >= total) return;
  int c = i % C; int t = i / C; int ox = t % OW; t /= OW; int oy = t % OH; int b = t / OH;
  float m = -3.4e38f;
  int iy0 = oy * 2 - 1, ix0 = ox * 2 - 1;
  for (int dy = 0; dy < 3; dy++) {
    int iy = iy0 + dy; if ((unsigned)iy >= (unsigned)H) continue;
    for (int dx = 0; dx < 3; dx++) {
      int ix = ix0 + dx; if ((unsigned)ix >= (unsigned)W) continue;
      m = fmaxf(m, bf2f(in[(((size_t)b * H + iy) * W + ix) * C + c]));
    }
  }
  out[i] = f2bf(m);
}

// ---------- FPN top-down: dst += nearest_2x_upsample(src), NHWC, in place ----------
__global__ void upsample_add(u16* __restrict__ dst, const u16* __restrict__ src,
                             int C, int H, int W, int SH, int SW, int N) {
  int i = blockIdx.x * blockDim.x + threadIdx.x;
  int total = N * C * H * W;
  if (i >= total) return;
  int c = i % C; int t = i / C; int x = t % W; t /= W; int y = t % H; int b = t / H;
  float v = bf2f(dst[i]) +
            bf2f(src[(((size_t)b * SH + (y >> 1)) * SW + (x >> 1)) * C + c]);
  dst[i] = f2bf(v);
}

// ---------- P6 = x[:, :, ::2, ::2] (NHWC) ----------
__global__ void subsample2(const u16* __restrict__ in, u16* __restrict__ out,
                           int C, int H, int W, int OH, int OW, int N) {
  int i = blockIdx.x * blockDim.x + threadIdx.x;
  int total = N * C * OH * OW;
  if (i >= total) return;
  int c = i % C; int t = i / C; int ox = t % OW; t /= OW; int oy = t % OH; int b = t / OH;
  out[i] = in[(((size_t)b * H + oy * 2) * W + ox * 2) * C + c];
}

// =====================================================================
extern "C" void kernel_launch(void* const* d_in, const int* in_sizes, int n_in,
                              void* d_out, int out_size, void* d_ws, size_t ws_size,
                              hipStream_t stream) {
  (void)in_sizes; (void)n_in; (void)out_size; (void)ws_size;
  const int N = 2;
  int cursor = 0;
  auto F = [&]() { return (const float*)d_in[cursor++]; };

  char* base = (char*)d_ws;
  size_t off = 0;
  auto alloc = [&](size_t bytes) -> void* {
    void* p = base + off;
    off += (bytes + 255) & ~(size_t)255;
    return p;
  };

  struct PW { const u16* w; int Cinpad; };
  auto pack = [&](const float* src, int Cout, int Cin, int k) -> PW {
    int kk = k * k;
    int Cinpad = (Cin + 31) & ~31;
    int Opad = (Cout + 63) & ~63;       // all 4 M-subtiles always readable
    int total = Opad * kk * Cinpad;
    u16* dst = (u16*)alloc((size_t)total * sizeof(u16));
    hipLaunchKernelGGL(pack_weight_bf16, dim3((total + 255) / 256), dim3(256), 0, stream,
                       src, dst, Cout, Cin, kk, Cinpad, total);
    return PW{dst, Cinpad};
  };

  auto conv = [&](const u16* in, PW w, int Cin, int k, int stride, int pad,
                  const float* scale, const float* shift, const u16* resid, bool relu,
                  int Cout, int H, int W, u16* out_bf, float* outf,
                  long long fbase, long long fbstride, int& OH, int& OW) {
    OH = (H + 2 * pad - k) / stride + 1;
    OW = (W + 2 * pad - k) / stride + 1;
    dim3 g((OH * OW + 127) / 128, (Cout + 63) / 64, N);
    hipLaunchKernelGGL(conv_wmma_bf16, g, dim3(128), 0, stream,
                       in, w.w, scale, shift, resid, relu ? 1 : 0,
                       out_bf, outf, fbase, fbstride,
                       Cin, H, W, Cout, k, stride, pad, OH, OW, w.Cinpad);
  };

  // -------- parse inputs (setup_inputs insertion order) + pack all weights --------
  const float* X = F();
  const float* stem_wf = F(); const float* stem_s = F(); const float* stem_b = F();
  PW stem_w = pack(stem_wf, 64, 3, 7);

  struct Blk {
    PW w1, w2, w3, wd;
    const float *s1, *b1, *s2, *b2, *s3, *b3, *sd, *bd;
    bool ds; int cin, cmid, cout, stride;
  };
  Blk blks[16]; int nb = 0; int layerStart[4];
  struct { int n, cin, cmid, cout, stride; } cfg[4] = {
      {3, 64, 64, 256, 1}, {4, 256, 128, 512, 2},
      {6, 512, 256, 1024, 2}, {3, 1024, 512, 2048, 2}};
  for (int L = 0; L < 4; L++) {
    layerStart[L] = nb;
    for (int i = 0; i < cfg[L].n; i++) {
      Blk bk{};
      bk.cin = (i == 0) ? cfg[L].cin : cfg[L].cout;
      bk.cmid = cfg[L].cmid; bk.cout = cfg[L].cout;
      bk.stride = (i == 0) ? cfg[L].stride : 1;
      bk.ds = (i == 0);
      bk.w1 = pack(F(), bk.cmid, bk.cin, 1);
      bk.w2 = pack(F(), bk.cmid, bk.cmid, 3);
      bk.w3 = pack(F(), bk.cout, bk.cmid, 1);
      bk.s1 = F(); bk.b1 = F(); bk.s2 = F(); bk.b2 = F(); bk.s3 = F(); bk.b3 = F();
      if (bk.ds) { bk.wd = pack(F(), bk.cout, bk.cin, 1); bk.sd = F(); bk.bd = F(); }
      blks[nb++] = bk;
    }
  }
  PW lat_w[4]; const float* lat_b[4]; PW fpn_w[4]; const float* fpn_b[4];
  int fc[4] = {256, 512, 1024, 2048};
  for (int i = 0; i < 4; i++) {
    lat_w[i] = pack(F(), 256, fc[i], 1); lat_b[i] = F();
    fpn_w[i] = pack(F(), 256, 256, 3);   fpn_b[i] = F();
  }
  PW rpn_w1 = pack(F(), 256, 256, 3); const float* rpn_b1 = F();
  PW rpn_w2 = pack(F(), 256, 256, 3); const float* rpn_b2 = F();
  PW cls_w  = pack(F(), 3, 256, 1);   const float* cls_b  = F();
  PW box_w  = pack(F(), 12, 256, 1);  const float* box_b  = F();

  // -------- activation workspace (bf16, NHWC) --------
  const size_t SLAB = 20480000;                   // max of 2*64*400^2, 2*256*200^2
  u16* xb    = (u16*)alloc((size_t)N * 3 * 800 * 800 * 2);
  u16* slabA = (u16*)alloc(SLAB * 2);
  u16* slabB = (u16*)alloc(SLAB * 2);
  u16* t1    = (u16*)alloc((size_t)10240000 * 2);
  u16* t2    = (u16*)alloc((size_t)10240000 * 2);
  u16* scb   = (u16*)alloc(SLAB * 2);
  size_t featsz[4] = {20480000, 10240000, 5120000, 2560000};
  size_t latsz[4]  = {20480000, 5120000, 1280000, 320000};
  u16 *feat[4], *latb[4], *outb[4];
  for (int i = 0; i < 4; i++) feat[i] = (u16*)alloc(featsz[i] * 2);
  for (int i = 0; i < 4; i++) latb[i] = (u16*)alloc(latsz[i] * 2);
  for (int i = 0; i < 4; i++) outb[i] = (u16*)alloc(latsz[i] * 2);
  u16* p6  = (u16*)alloc((size_t)N * 256 * 13 * 13 * 2);
  u16* rt1 = (u16*)alloc(SLAB * 2);
  u16* rt2 = (u16*)alloc(SLAB * 2);

  // -------- stem --------
  { int n = N * 3 * 800 * 800;
    hipLaunchKernelGGL(cvt_nchw_to_nhwc_bf16, dim3((n + 255) / 256), dim3(256), 0, stream,
                       X, xb, 3, 800 * 800, n); }
  int OH, OW;
  conv(xb, stem_w, 3, 7, 2, 3, stem_s, stem_b, nullptr, true, 64, 800, 800,
       slabA, nullptr, 0, 0, OH, OW);                           // -> 400x400
  { int total = N * 64 * 200 * 200;
    hipLaunchKernelGGL(maxpool3s2, dim3((total + 255) / 256), dim3(256), 0, stream,
                       slabA, slabB, 64, 400, 400, 200, 200, N); }

  // -------- ResNet trunk --------
  u16* cur = slabB; int H = 200, W = 200;
  for (int L = 0; L < 4; L++) {
    for (int i = 0; i < cfg[L].n; i++) {
      Blk& bk = blks[layerStart[L] + i];
      int h1H, h1W, h2H, h2W, dH, dW;
      conv(cur, bk.w1, bk.cin, 1, 1, 0, bk.s1, bk.b1, nullptr, true,
           bk.cmid, H, W, t1, nullptr, 0, 0, h1H, h1W);
      conv(t1, bk.w2, bk.cmid, 3, bk.stride, 1, bk.s2, bk.b2, nullptr, true,
           bk.cmid, h1H, h1W, t2, nullptr, 0, 0, h2H, h2W);
      const u16* resid = cur;
      if (bk.ds) {
        conv(cur, bk.wd, bk.cin, 1, bk.stride, 0, bk.sd, bk.bd, nullptr, false,
             bk.cout, H, W, scb, nullptr, 0, 0, dH, dW);
        resid = scb;
      }
      bool last = (i == cfg[L].n - 1);
      u16* target = last ? feat[L] : ((cur == slabA) ? slabB : slabA);
      conv(t2, bk.w3, bk.cmid, 1, 1, 0, bk.s3, bk.b3, resid, true,
           bk.cout, h2H, h2W, target, nullptr, 0, 0, H, W);
      cur = target;
    }
  }

  // -------- FPN --------
  int fH[4] = {200, 100, 50, 25};
  for (int i = 0; i < 4; i++) {
    int oh, ow;
    conv(feat[i], lat_w[i], fc[i], 1, 1, 0, nullptr, lat_b[i], nullptr, false,
         256, fH[i], fH[i], latb[i], nullptr, 0, 0, oh, ow);
  }
  for (int i = 2; i >= 0; i--) {
    int total = N * 256 * fH[i] * fH[i];
    hipLaunchKernelGGL(upsample_add, dim3((total + 255) / 256), dim3(256), 0, stream,
                       latb[i], latb[i + 1], 256, fH[i], fH[i], fH[i + 1], fH[i + 1], N);
  }
  for (int i = 0; i < 4; i++) {
    int oh, ow;
    conv(latb[i], fpn_w[i], 256, 3, 1, 1, nullptr, fpn_b[i], nullptr, false,
         256, fH[i], fH[i], outb[i], nullptr, 0, 0, oh, ow);
  }
  { int total = N * 256 * 13 * 13;
    hipLaunchKernelGGL(subsample2, dim3((total + 255) / 256), dim3(256), 0, stream,
                       outb[3], p6, 256, 25, 25, 13, 13, N); }

  // -------- RPN head on P2..P6, writing fp32 straight into d_out concat layout --------
  const u16* levels[5] = {outb[0], outb[1], outb[2], outb[3], p6};
  int lH[5] = {200, 100, 50, 25, 13};
  long long obj_off = 0;
  const long long BSTRIDE = 799410;
  float* OUT = (float*)d_out;
  for (int l = 0; l < 5; l++) {
    int oh, ow;
    long long HWl = (long long)lH[l] * lH[l];
    conv(levels[l], rpn_w1, 256, 3, 1, 1, nullptr, rpn_b1, nullptr, true,
         256, lH[l], lH[l], rt1, nullptr, 0, 0, oh, ow);
    conv(rt1, rpn_w2, 256, 3, 1, 1, nullptr, rpn_b2, nullptr, true,
         256, lH[l], lH[l], rt2, nullptr, 0, 0, oh, ow);
    conv(rt2, cls_w, 256, 1, 1, 0, nullptr, cls_b, nullptr, false,
         3, lH[l], lH[l], nullptr, OUT, obj_off, BSTRIDE, oh, ow);
    conv(rt2, box_w, 256, 1, 1, 0, nullptr, box_b, nullptr, false,
         12, lH[l], lH[l], nullptr, OUT, obj_off + 3 * HWl, BSTRIDE, oh, ow);
    obj_off += 15 * HWl;
  }
}
